// TransformerDecoder_29910152249656
// MI455X (gfx1250) — matrix operand order repaired and checked by reading, not compiled, not run on hardware
//
#include <hip/hip_runtime.h>
#include <hip/hip_bf16.h>

typedef unsigned short u16;
typedef __bf16 bf16_t;
typedef bf16_t v16bf __attribute__((ext_vector_type(16)));
typedef float  v8f   __attribute__((ext_vector_type(8)));
typedef int    v4i   __attribute__((vector_size(16)));   // matches builtin param type

struct U32x8  { uint4 a, b; };
struct U16x16 { u16 s[16]; };

__device__ __forceinline__ u16 f2bf(float f) {
    unsigned u = __builtin_bit_cast(unsigned, f);
    unsigned r = u + 0x7FFFu + ((u >> 16) & 1u);   // round-to-nearest-even
    return (u16)(r >> 16);
}

__device__ __forceinline__ v16bf mkfrag(uint4 a, uint4 b) {
    U32x8 t{a, b};
    return __builtin_bit_cast(v16bf, t);
}

__device__ __forceinline__ v16bf pack16(float4 a, float4 b, float4 c, float4 d) {
    U16x16 t;
    t.s[0]=f2bf(a.x);  t.s[1]=f2bf(a.y);  t.s[2]=f2bf(a.z);  t.s[3]=f2bf(a.w);
    t.s[4]=f2bf(b.x);  t.s[5]=f2bf(b.y);  t.s[6]=f2bf(b.z);  t.s[7]=f2bf(b.w);
    t.s[8]=f2bf(c.x);  t.s[9]=f2bf(c.y);  t.s[10]=f2bf(c.z); t.s[11]=f2bf(c.w);
    t.s[12]=f2bf(d.x); t.s[13]=f2bf(d.y); t.s[14]=f2bf(d.z); t.s[15]=f2bf(d.w);
    return __builtin_bit_cast(v16bf, t);
}

__device__ __forceinline__ v8f wmma_bf16(v16bf a, v16bf b, v8f c) {
    // 8 args: (neg_a, A, neg_b, B, c_mod, C, reuse_a, reuse_b)
    return __builtin_amdgcn_wmma_f32_16x16x32_bf16(false, a, false, b, (short)0, c, false, false);
}

// --- CDNA5 async memory->LDS copy (ASYNCcnt-tracked), with sync fallback ----
#if defined(__AMDGCN__) && __has_builtin(__builtin_amdgcn_global_load_async_to_lds_b128)
#define HAVE_ASYNC_LDS 1
#else
#define HAVE_ASYNC_LDS 0
#endif

__device__ __forceinline__ void cp16_g2l(const float* g, float* l) {
#if HAVE_ASYNC_LDS
    __builtin_amdgcn_global_load_async_to_lds_b128((v4i*)g, (v4i*)l, 0, 0);
#else
    *(float4*)l = *(const float4*)g;
#endif
}

template <int N>
__device__ __forceinline__ void wait_async() {
#if defined(__AMDGCN__) && __has_builtin(__builtin_amdgcn_s_wait_asynccnt)
    __builtin_amdgcn_s_wait_asynccnt(N);
#elif defined(__AMDGCN__)
    if (N == 0) asm volatile("s_wait_asynccnt 0" ::: "memory");
    else        asm volatile("s_wait_asynccnt 4" ::: "memory");
#endif
}

// ---------------------------------------------------------------------------
// Embedding: h[t,d] = tok_emb[x[t], d] + pos_emb[t, d]
// ---------------------------------------------------------------------------
__global__ __launch_bounds__(256) void embed_kernel(
    const int* __restrict__ x, const float* __restrict__ tok,
    const float* __restrict__ pos, float* __restrict__ out)
{
    const int t = blockIdx.x;
    const int d = threadIdx.x * 4;
    const long id = x[t];
    float4 a = *(const float4*)(tok + id * 1024 + d);
    float4 p = *(const float4*)(pos + (long)t * 1024 + d);
    float4 o; o.x = a.x + p.x; o.y = a.y + p.y; o.z = a.z + p.z; o.w = a.w + p.w;
    *(float4*)(out + (long)t * 1024 + d) = o;
}

// ---------------------------------------------------------------------------
// LayerNorm over D=1024: one block per row, 256 threads * float4
// ---------------------------------------------------------------------------
__global__ __launch_bounds__(256) void ln_kernel(
    const float* __restrict__ x, const float* __restrict__ g,
    const float* __restrict__ b, float* __restrict__ out)
{
    __shared__ float red[256];
    const int t = blockIdx.x, tid = threadIdx.x;
    const float* xr = x + (long)t * 1024;
    float4 vv = *(const float4*)(xr + tid * 4);

    red[tid] = vv.x + vv.y + vv.z + vv.w;
    __syncthreads();
    for (int o = 128; o > 0; o >>= 1) { if (tid < o) red[tid] += red[tid + o]; __syncthreads(); }
    const float mean = red[0] * (1.f / 1024.f);
    __syncthreads();

    float dx = vv.x - mean, dy = vv.y - mean, dz = vv.z - mean, dw = vv.w - mean;
    red[tid] = dx * dx + dy * dy + dz * dz + dw * dw;
    __syncthreads();
    for (int o = 128; o > 0; o >>= 1) { if (tid < o) red[tid] += red[tid + o]; __syncthreads(); }
    const float inv = rsqrtf(red[0] * (1.f / 1024.f) + 1e-5f);

    float4 gg = *(const float4*)(g + tid * 4);
    float4 bb = *(const float4*)(b + tid * 4);
    float4 ov;
    ov.x = dx * inv * gg.x + bb.x;
    ov.y = dy * inv * gg.y + bb.y;
    ov.z = dz * inv * gg.z + bb.z;
    ov.w = dw * inv * gg.w + bb.w;
    *(float4*)(out + (long)t * 1024 + tid * 4) = ov;
}

// ---------------------------------------------------------------------------
// Generic WMMA GEMM: C[M,N] = act(A[M,K] x B[K,N] + bias[N]) (+res)
// BM=128, BN=64, BK=32; 256 threads = 8 waves; wave w -> rows [w*16, w*16+16).
// A tiles: fp32, DMA'd global->LDS asynchronously (double buffered),
//          converted to bf16 in registers when building fragments.
// B tiles: fp32 global -> bf16 transposed in LDS (double buffered).
// grid.z batches over heads via zs* strides. All dims multiples of tiles.
// ---------------------------------------------------------------------------
__global__ __launch_bounds__(256) void gemm_kernel(
    const float* __restrict__ A, long lda, long zsA,
    const float* __restrict__ B, long ldb, long zsB,
    float* C, long ldc, long zsC,
    const float* __restrict__ bias, long zsBias,
    const float* res, long zsRes,
    int M, int N, int K, int relu)
{
    (void)M; (void)N;
    __shared__ __attribute__((aligned(16))) float Afp[2][128][36]; // fp32, padded rows
    __shared__ __attribute__((aligned(16))) u16   Bt[2][64][40];   // bf16, Bt[n][k], padded

    const long bz = blockIdx.z;
    A += bz * zsA; B += bz * zsB; C += bz * zsC;
    if (bias) bias += bz * zsBias;
    if (res)  res  += bz * zsRes;

    const int mBase = blockIdx.y * 128, nBase = blockIdx.x * 64;
    const int tid = threadIdx.x, w = tid >> 5, L = tid & 31, lh = L >> 4, lr = L & 15;

    v8f acc[4] = {};

    // ---- tile staging helpers (lambdas keep per-thread index math in one place)
    auto stageA = [&](int buf, int k0) {   // 4 async b128 issues per lane
#pragma unroll
        for (int p = 0; p < 4; ++p) {
            const int row = p * 32 + (tid >> 3), col = (tid & 7) * 4;
            cp16_g2l(A + (long)(mBase + row) * lda + k0 + col, &Afp[buf][row][col]);
        }
    };
    auto stageB = [&](int buf, int k0) {   // sync: load fp32, convert, transpose
#pragma unroll
        for (int p = 0; p < 2; ++p) {
            const int kr = p * 16 + (tid >> 4), nc = (tid & 15) * 4;
            float4 fb = *(const float4*)(B + (long)(k0 + kr) * ldb + nBase + nc);
            Bt[buf][nc + 0][kr] = f2bf(fb.x); Bt[buf][nc + 1][kr] = f2bf(fb.y);
            Bt[buf][nc + 2][kr] = f2bf(fb.z); Bt[buf][nc + 3][kr] = f2bf(fb.w);
        }
    };

    stageA(0, 0);
    stageB(0, 0);

    for (int k0 = 0; k0 < K; k0 += 32) {
        const int buf = (k0 >> 5) & 1;
        const bool more = (k0 + 32) < K;
        if (more) {                        // prefetch next tile into other buffer
            stageA(buf ^ 1, k0 + 32);
            stageB(buf ^ 1, k0 + 32);
            wait_async<4>();               // retire current buffer's 4 async loads
        } else {
            wait_async<0>();
        }
        __syncthreads();

        // A fragment: lane lr -> row, lane-half K split {lh*8+0..7, 16+lh*8+0..7}
        const float* arow = &Afp[buf][w * 16 + lr][0];
        float4 a0 = *(const float4*)(arow + lh * 8);
        float4 a1 = *(const float4*)(arow + lh * 8 + 4);
        float4 a2 = *(const float4*)(arow + 16 + lh * 8);
        float4 a3 = *(const float4*)(arow + 16 + lh * 8 + 4);
        v16bf af = pack16(a0, a1, a2, a3);

#pragma unroll
        for (int j = 0; j < 4; ++j) {
            // B fragment: lane lr -> col n, lane-half K split {lh*8+0..7, 16+lh*8+0..7}
            const uint4* br = (const uint4*)(&Bt[buf][j * 16 + lr][0]);
            v16bf bf_ = mkfrag(br[lh], br[2 + lh]);
            acc[j] = wmma_bf16(af, bf_, acc[j]);
        }
        __syncthreads();                   // everyone done reading buf before reuse
    }

#pragma unroll
    for (int j = 0; j < 4; ++j) {
        const int n = nBase + j * 16 + lr;
        const float bv = bias ? bias[n] : 0.f;
#pragma unroll
        for (int i = 0; i < 8; ++i) {
            const int m = mBase + w * 16 + i + 8 * lh;   // C layout: VGPR i, lane-half
            float val = acc[j][i] + bv;
            if (relu) val = fmaxf(val, 0.f);
            const long off = (long)m * ldc + n;
            if (res) val += res[off];
            C[off] = val;
        }
    }
}

// ---------------------------------------------------------------------------
// Causal flash attention, one head per blockIdx.y, 128 q-rows per block,
// each wave owns 16 q-rows and streams 32-key tiles. h += softmax(qk^T)*v
// ---------------------------------------------------------------------------
__global__ __launch_bounds__(256) void attn_kernel(
    const float* __restrict__ q, const float* __restrict__ k,
    const float* __restrict__ v, float* hbuf, int T, float scale)
{
    __shared__ __attribute__((aligned(16))) u16 P[8][16][40];   // per-wave probs, padded

    const int head = blockIdx.y;
    const int tid = threadIdx.x, w = tid >> 5, L = tid & 31, lh = L >> 4, lr = L & 15;
    const int qb = blockIdx.x * 128 + w * 16;

    const float* qh = q + (long)head * T * 64;
    const float* kh = k + (long)head * T * 64;
    const float* vh = v + (long)head * T * 64;

    // Q fragments for K(dim)=64 as two 32-chunks (loaded once)
    const float* qp = qh + (long)(qb + lr) * 64;
    v16bf aq[2];
#pragma unroll
    for (int c = 0; c < 2; ++c) {
        float4 r0 = *(const float4*)(qp + c * 32 + lh * 16);
        float4 r1 = *(const float4*)(qp + c * 32 + lh * 16 + 4);
        float4 r2 = *(const float4*)(qp + c * 32 + lh * 16 + 8);
        float4 r3 = *(const float4*)(qp + c * 32 + lh * 16 + 12);
        aq[c] = pack16(r0, r1, r2, r3);
    }

    float m[8], l[8];
#pragma unroll
    for (int i = 0; i < 8; ++i) { m[i] = -1e30f; l[i] = 0.f; }
    v8f o[4] = {};

    for (int s = 0; s < qb + 16; s += 32) {
        // scores 16x32 = two 16x16 tiles, K(dim)=64 via two chunks
        v8f sc0 = {}, sc1 = {};
#pragma unroll
        for (int c = 0; c < 2; ++c) {
            const float* kp0 = kh + (long)(s + lr) * 64 + c * 32 + lh * 16;
            v16bf kb0 = pack16(((const float4*)kp0)[0], ((const float4*)kp0)[1],
                               ((const float4*)kp0)[2], ((const float4*)kp0)[3]);
            const float* kp1 = kh + (long)(s + 16 + lr) * 64 + c * 32 + lh * 16;
            v16bf kb1 = pack16(((const float4*)kp1)[0], ((const float4*)kp1)[1],
                               ((const float4*)kp1)[2], ((const float4*)kp1)[3]);
            sc0 = wmma_bf16(aq[c], kb0, sc0);
            sc1 = wmma_bf16(aq[c], kb1, sc1);
        }

        // causal mask + online softmax (16-lane row reductions)
#pragma unroll
        for (int i = 0; i < 8; ++i) {
            const int row = qb + i + 8 * lh;
            float x0 = sc0[i] * scale; if (s + lr > row)      x0 = -1e30f;
            float x1 = sc1[i] * scale; if (s + 16 + lr > row) x1 = -1e30f;
            float rm = fmaxf(x0, x1);
#pragma unroll
            for (int off = 8; off > 0; off >>= 1) rm = fmaxf(rm, __shfl_xor(rm, off, 16));
            const float mn = fmaxf(m[i], rm);
            const float corr = __expf(m[i] - mn);
            const float p0 = __expf(x0 - mn), p1 = __expf(x1 - mn);
            float rs = p0 + p1;
#pragma unroll
            for (int off = 8; off > 0; off >>= 1) rs += __shfl_xor(rs, off, 16);
            l[i] = l[i] * corr + rs;
            m[i] = mn;
#pragma unroll
            for (int j = 0; j < 4; ++j) o[j][i] *= corr;
            P[w][i + 8 * lh][lr]      = f2bf(p0);
            P[w][i + 8 * lh][16 + lr] = f2bf(p1);
        }
        asm volatile("" ::: "memory");   // keep LDS write->read order (intra-wave, in-order DS)

        // probs A-fragment from LDS (re-layout C->A)
        const uint4* pr = (const uint4*)(&P[w][lr][0]);
        v16bf ap = mkfrag(pr[2 * lh], pr[2 * lh + 1]);

        // o += probs(16x32) x V(32x64), four 16-col tiles
#pragma unroll
        for (int j = 0; j < 4; ++j) {
            U16x16 tv;
#pragma unroll
            for (int e = 0; e < 16; ++e)
                tv.s[e] = f2bf(vh[(long)(s + lh * 16 + e) * 64 + j * 16 + lr]);
            o[j] = wmma_bf16(ap, __builtin_bit_cast(v16bf, tv), o[j]);
        }
    }

    // h += o / l  (head concat: cols head*64 + d)
#pragma unroll
    for (int j = 0; j < 4; ++j) {
        const int d = j * 16 + lr;
#pragma unroll
        for (int i = 0; i < 8; ++i) {
            const int row = qb + i + 8 * lh;
            const long off = (long)row * 1024 + head * 64 + d;
            hbuf[off] += o[j][i] / l[i];
        }
    }
}

// ---------------------------------------------------------------------------
extern "C" void kernel_launch(void* const* d_in, const int* in_sizes, int n_in,
                              void* d_out, int out_size, void* d_ws, size_t ws_size,
                              hipStream_t stream)
{
    (void)in_sizes; (void)n_in; (void)out_size; (void)ws_size;

    const int*   x    = (const int*)d_in[0];
    const float* tok  = (const float*)d_in[1];
    const float* pos  = (const float*)d_in[2];
    const float* wq   = (const float*)d_in[3];
    const float* bq   = (const float*)d_in[4];
    const float* wk   = (const float*)d_in[5];
    const float* bk   = (const float*)d_in[6];
    const float* wv   = (const float*)d_in[7];
    const float* bv   = (const float*)d_in[8];
    const float* ln1g = (const float*)d_in[9];
    const float* ln1b = (const float*)d_in[10];
    const float* ln2g = (const float*)d_in[11];
    const float* ln2b = (const float*)d_in[12];
    const float* w1   = (const float*)d_in[13];
    const float* b1   = (const float*)d_in[14];
    const float* w2   = (const float*)d_in[15];
    const float* b2   = (const float*)d_in[16];
    const float* lnfg = (const float*)d_in[17];
    const float* lnfb = (const float*)d_in[18];
    const float* wvoc = (const float*)d_in[19];
    const float* bvoc = (const float*)d_in[20];
    float* out = (float*)d_out;

    const long TD = 2048l * 1024;           // 2M floats
    float* hB  = (float*)d_ws;              // [2048,1024] residual stream
    float* xnB = hB  + TD;                  // [2048,1024] LN output
    float* qB  = xnB + TD;                  // [16,2048,64]
    float* kB  = qB  + TD;
    float* vB  = kB  + TD;
    float* ffB = vB  + TD;                  // [2048,4096]

    embed_kernel<<<2048, 256, 0, stream>>>(x, tok, pos, hB);

    const float scale = 0.03125f;           // 1/sqrt(D=1024), as in reference

    for (int lyr = 0; lyr < 4; ++lyr) {
        ln_kernel<<<2048, 256, 0, stream>>>(hB, ln1g + lyr * 1024, ln1b + lyr * 1024, xnB);

        // QKV projections: grid.z = head; B = w[l,h] is [1024,64], bias per head
        const long wOff = (long)lyr * 16 * 1024 * 64, bOff = (long)lyr * 16 * 64;
        gemm_kernel<<<dim3(1, 16, 16), 256, 0, stream>>>(
            xnB, 1024, 0, wq + wOff, 64, 1024l * 64, qB, 64, 2048l * 64,
            bq + bOff, 64, nullptr, 0, 2048, 64, 1024, 0);
        gemm_kernel<<<dim3(1, 16, 16), 256, 0, stream>>>(
            xnB, 1024, 0, wk + wOff, 64, 1024l * 64, kB, 64, 2048l * 64,
            bk + bOff, 64, nullptr, 0, 2048, 64, 1024, 0);
        gemm_kernel<<<dim3(1, 16, 16), 256, 0, stream>>>(
            xnB, 1024, 0, wv + wOff, 64, 1024l * 64, vB, 64, 2048l * 64,
            bv + bOff, 64, nullptr, 0, 2048, 64, 1024, 0);

        attn_kernel<<<dim3(16, 16), 256, 0, stream>>>(qB, kB, vB, hB, 2048, scale);

        ln_kernel<<<2048, 256, 0, stream>>>(hB, ln2g + lyr * 1024, ln2b + lyr * 1024, xnB);

        gemm_kernel<<<dim3(64, 16, 1), 256, 0, stream>>>(
            xnB, 1024, 0, w1 + (long)lyr * 1024 * 4096, 4096, 0, ffB, 4096, 0,
            b1 + (long)lyr * 4096, 0, nullptr, 0, 2048, 4096, 1024, 1 /*relu*/);

        gemm_kernel<<<dim3(16, 16, 1), 256, 0, stream>>>(
            ffB, 4096, 0, w2 + (long)lyr * 4096 * 1024, 1024, 0, hB, 1024, 0,
            b2 + (long)lyr * 1024, 0, hB /*residual*/, 0, 2048, 1024, 4096, 0);
    }

    ln_kernel<<<2048, 256, 0, stream>>>(hB, lnfg, lnfb, xnB);

    // logits = xn @ w_vocab + b_vocab  -> d_out [2048, 32000]
    gemm_kernel<<<dim3(500, 16, 1), 256, 0, stream>>>(
        xnB, 1024, 0, wvoc, 32000, 0, out, 32000, 0,
        bvoc, 0, nullptr, 0, 2048, 32000, 1024, 0);
}
